// RGAT_44564580663521
// MI455X (gfx1250) — matrix operand order repaired and checked by reading, not compiled, not run on hardware
//
#include <hip/hip_runtime.h>
#include <hip/hip_bf16.h>
#include <stdint.h>

#define N_NODES 100000
#define N_EDGES 800000
#define N_REL   3
#define F_DIM   128   // IN == H1*HID == OUT == 128

typedef __attribute__((ext_vector_type(16))) __bf16 bf16x16;
typedef __attribute__((ext_vector_type(8)))  float  f32x8;

union Frag { unsigned u[8]; bf16x16 v; };

__device__ __forceinline__ unsigned f32key(float x) {
  unsigned u = __float_as_uint(x);
  return (u & 0x80000000u) ? ~u : (u | 0x80000000u);
}
__device__ __forceinline__ float keyf32(unsigned k) {
  return (k & 0x80000000u) ? __uint_as_float(k & 0x7fffffffu) : __uint_as_float(~k);
}
__device__ __forceinline__ unsigned short f2bf(float f) {
  unsigned u = __float_as_uint(f);
  u += 0x7fffu + ((u >> 16) & 1u);   // round-to-nearest-even
  return (unsigned short)(u >> 16);
}
__device__ __forceinline__ float lrelu(float v) { return v > 0.f ? v : 0.2f * v; }

// ---------------- conversion / init kernels ----------------

__global__ void k_to_bf16(const float* __restrict__ in, unsigned short* __restrict__ out, int count) {
  int i = blockIdx.x * blockDim.x + threadIdx.x;
  if (i < count) out[i] = f2bf(in[i]);
}

__global__ void k_zero_f32(float* __restrict__ p, int count) {
  int i = blockIdx.x * blockDim.x + threadIdx.x;
  if (i < count) p[i] = 0.f;
}

__global__ void k_init_nodes(unsigned* __restrict__ nmaxKey, float* __restrict__ nsum, int count) {
  int i = blockIdx.x * blockDim.x + threadIdx.x;
  if (i < count) { nmaxKey[i] = 0x007FFFFFu; /* key(-inf) */ nsum[i] = 0.f; }
}

// ---------------- W pre-pack into WMMA B-fragment layout ----------------
// Per relation: [ct(8 col tiles)][kb(4 k blocks)][lane(32)][p(8 dwords)]
// dword p of lane holds bf16 pair (K=2p, K=2p+1), +16 K for lanes 16-31; col = ct*16 + (lane&15)
#define WPACK_DW_PER_REL (8 * 4 * 32 * 8)   // 8192 dwords

__global__ void k_pack_W(const float* __restrict__ W, unsigned* __restrict__ Bp, int total) {
  int i = blockIdx.x * blockDim.x + threadIdx.x;
  if (i >= total) return;
  int p    = i & 7;
  int lane = (i >> 3) & 31;
  int kb   = (i >> 8) & 3;
  int ct   = (i >> 10) & 7;
  int r    = i >> 13;
  int laneLo = lane & 15, laneHi = lane >> 4;
  int col = ct * 16 + laneLo;
  int k   = kb * 32 + (p << 1) + (laneHi << 4);
  const float* Wr = W + (size_t)r * F_DIM * F_DIM;
  unsigned lo = f2bf(Wr[(size_t)k * F_DIM + col]);
  unsigned hi = f2bf(Wr[(size_t)(k + 1) * F_DIM + col]);
  Bp[i] = lo | (hi << 16);
}

// ---------------- WMMA GEMM: C[M,128] = A[M,128] * B[128,128] ----------------
// block = 8 waves; block -> 16-row M tile, wave w -> 16-col N tile.
// A: row-major bf16 (raw ushort). Bp: pre-packed fragment layout (see k_pack_W).
__global__ __launch_bounds__(256, 1) void k_gemm_bf16(
    const unsigned short* __restrict__ A, const unsigned* __restrict__ Bp,
    float* __restrict__ C)
{
  const int lane   = threadIdx.x & 31;
  const int wave   = threadIdx.x >> 5;     // column tile 0..7
  const int laneLo = lane & 15;
  const int laneHi = lane >> 4;
  const int row0   = blockIdx.x * 16;
  const int col0   = wave * 16;

  f32x8 c = {};
  #pragma unroll
  for (int kb = 0; kb < 4; ++kb) {
    Frag a, b;
    // A fragment: 16x32 bf16, lane = row (mod 16); dword p holds K pair
    const unsigned short* ap = A + (size_t)(row0 + laneLo) * F_DIM + kb * 32;
    #pragma unroll
    for (int p = 0; p < 8; ++p) {
      int k0 = (p << 1) + (laneHi << 4);
      a.u[p] = *(const unsigned*)(ap + k0);   // merges into b128 loads
    }
    // B fragment: 2 contiguous 16B loads from packed layout
    const unsigned* bb = Bp + (((size_t)wave * 4 + kb) * 32 + lane) * 8;
    *(uint4*)&b.u[0] = *(const uint4*)bb;
    *(uint4*)&b.u[4] = *(const uint4*)(bb + 4);

    c = __builtin_amdgcn_wmma_f32_16x16x32_bf16(false, a.v, false, b.v,
                                                (short)0, c, false, false);
  }
  // D layout: element j -> row j + 8*laneHi, col lane&15
  float* cp = C + (size_t)row0 * F_DIM + col0 + laneLo;
  #pragma unroll
  for (int j = 0; j < 8; ++j)
    cp[(size_t)(j + (laneHi << 3)) * F_DIM] = c[j];
}

// ---------------- attention projections: el/er [N,H] ----------------
// wave per (node, head): coalesced loads + shfl butterfly reduction
__global__ __launch_bounds__(256) void k_attn_proj(
    const float* __restrict__ feat,
    const float* __restrict__ al, const float* __restrict__ ar,
    float* __restrict__ el, float* __restrict__ er,
    int H, int D)
{
  int lane = threadIdx.x & 31;
  int w = blockIdx.x * (blockDim.x >> 5) + (threadIdx.x >> 5);   // n*H + h
  if (w >= N_NODES * H) return;
  int n = w / H, h = w - n * H;
  const float* f   = feat + (size_t)n * F_DIM + h * D;
  const float* alh = al + h * D;
  const float* arh = ar + h * D;
  float sl = 0.f, sr = 0.f;
  for (int d = lane; d < D; d += 32) {
    float v = f[d];
    sl += v * alh[d];
    sr += v * arh[d];
  }
  #pragma unroll
  for (int off = 16; off; off >>= 1) {
    sl += __shfl_down(sl, off, 32);
    sr += __shfl_down(sr, off, 32);
  }
  if (lane == 0) { el[w] = sl; er[w] = sr; }
}

// ---------------- edge softmax passes ----------------

__global__ void k_edge_max(const int* __restrict__ src, const int* __restrict__ dst,
                           const float* __restrict__ el, const float* __restrict__ er,
                           unsigned* __restrict__ nmaxKey, int H)
{
  int i = blockIdx.x * blockDim.x + threadIdx.x;   // e*H + h
  if (i >= N_EDGES * H) return;
  int e = i / H, h = i - e * H;
  int s = src[e], t = dst[e];
  float v = lrelu(el[s * H + h] + er[t * H + h]);
  atomicMax(&nmaxKey[t * H + h], f32key(v));
}

__global__ void k_edge_sum(const int* __restrict__ src, const int* __restrict__ dst,
                           const float* __restrict__ el, const float* __restrict__ er,
                           const unsigned* __restrict__ nmaxKey, float* __restrict__ nsum, int H)
{
  int i = blockIdx.x * blockDim.x + threadIdx.x;
  if (i >= N_EDGES * H) return;
  int e = i / H, h = i - e * H;
  int s = src[e], t = dst[e];
  float v = lrelu(el[s * H + h] + er[t * H + h]);
  float m = keyf32(nmaxKey[t * H + h]);
  atomicAdd(&nsum[t * H + h], __expf(v - m));
}

// wave per edge; lane handles channels c = lane + 32q, q=0..3
__global__ __launch_bounds__(256, 1) void k_edge_aggregate(
    const int* __restrict__ src, const int* __restrict__ dst,
    const float* __restrict__ el, const float* __restrict__ er,
    const unsigned* __restrict__ nmaxKey, const float* __restrict__ nsum,
    const float* __restrict__ feat, float* __restrict__ acc,
    int H, int dshift)
{
  int lane = threadIdx.x & 31;
  int e = blockIdx.x * (blockDim.x >> 5) + (threadIdx.x >> 5);
  if (e >= N_EDGES) return;
  int s = src[e], t = dst[e];
  const float* fs = feat + (size_t)s * F_DIM;
  float* at = acc + (size_t)t * F_DIM;
  #pragma unroll
  for (int q = 0; q < 4; ++q) {
    int c = lane + (q << 5);
    int h = c >> dshift;
    float v  = lrelu(el[s * H + h] + er[t * H + h]);
    float m  = keyf32(nmaxKey[t * H + h]);
    float dn = nsum[t * H + h]; dn = dn > 1e-9f ? dn : 1e-9f;
    float a  = __expf(v - m) / dn;
    atomicAdd(&at[c], a * fs[c]);
  }
}

// ---------------- layer epilogues ----------------

__global__ void k_finalize1(const float* __restrict__ acc, const float* __restrict__ x,
                            const float* __restrict__ b1,
                            float* __restrict__ h, unsigned short* __restrict__ hb)
{
  int i = blockIdx.x * blockDim.x + threadIdx.x;   // n*F + c
  if (i >= N_NODES * F_DIM) return;
  int c = i & (F_DIM - 1);
  float bs = (b1[c] + b1[F_DIM + c] + b1[2 * F_DIM + c]) * (1.f / 3.f);
  float v = acc[i] * (1.f / 3.f) + x[i] + bs;
  v = v > 0.f ? v : (__expf(v) - 1.f);   // ELU
  h[i] = v;
  hb[i] = f2bf(v);
}

__global__ void k_finalize2(const float* __restrict__ acc, const float* __restrict__ h,
                            const float* __restrict__ b2, float* __restrict__ out)
{
  int i = blockIdx.x * blockDim.x + threadIdx.x;
  if (i >= N_NODES * F_DIM) return;
  int c = i & (F_DIM - 1);
  float bs = (b2[c] + b2[F_DIM + c] + b2[2 * F_DIM + c]) * (1.f / 3.f);
  out[i] = acc[i] * (1.f / 3.f) + h[i] + bs;
}

// ---------------- host launcher ----------------

static inline int cdiv(long long a, int b) { return (int)((a + b - 1) / b); }

extern "C" void kernel_launch(void* const* d_in, const int* in_sizes, int n_in,
                              void* d_out, int out_size, void* d_ws, size_t ws_size,
                              hipStream_t stream) {
  (void)in_sizes; (void)n_in; (void)out_size; (void)ws_size;
  const float* x        = (const float*)d_in[0];
  const int*   edge_src = (const int*)d_in[1];   // [R,E]
  const int*   edge_dst = (const int*)d_in[2];   // [R,E]
  const float* W1  = (const float*)d_in[3];      // [R,128,128]
  const float* b1  = (const float*)d_in[4];      // [R,128]
  const float* al1 = (const float*)d_in[5];      // [R,4,32]
  const float* ar1 = (const float*)d_in[6];
  const float* W2  = (const float*)d_in[7];      // [R,128,128]
  const float* b2  = (const float*)d_in[8];
  const float* al2 = (const float*)d_in[9];      // [R,1,128]
  const float* ar2 = (const float*)d_in[10];
  float* out = (float*)d_out;

  // workspace carve
  char* p = (char*)d_ws;
  auto take = [&](size_t bytes) { char* r = p; p += (bytes + 255) & ~(size_t)255; return r; };
  unsigned short* hb  = (unsigned short*)take((size_t)N_NODES * F_DIM * 2);
  unsigned* W1p = (unsigned*)take((size_t)N_REL * WPACK_DW_PER_REL * 4);
  unsigned* W2p = (unsigned*)take((size_t)N_REL * WPACK_DW_PER_REL * 4);
  float*    feat = (float*)take((size_t)N_NODES * F_DIM * 4);
  float*    el   = (float*)take((size_t)N_NODES * 4 * 4);
  float*    er   = (float*)take((size_t)N_NODES * 4 * 4);
  unsigned* nmax = (unsigned*)take((size_t)N_NODES * 4 * 4);
  float*    nsum = (float*)take((size_t)N_NODES * 4 * 4);
  float*    acc  = (float*)take((size_t)N_NODES * F_DIM * 4);
  float*    h    = (float*)take((size_t)N_NODES * F_DIM * 4);

  const int NT = 256;
  const int NF = N_NODES * F_DIM;
  const int packTotal = N_REL * WPACK_DW_PER_REL;

  k_to_bf16<<<cdiv(NF, NT), NT, 0, stream>>>(x, hb, NF);
  k_pack_W<<<cdiv(packTotal, NT), NT, 0, stream>>>(W1, W1p, packTotal);
  k_pack_W<<<cdiv(packTotal, NT), NT, 0, stream>>>(W2, W2p, packTotal);

  const int gemmGrid = N_NODES / 16;        // 6250
  const int aggGrid  = cdiv((long long)N_EDGES * 32, NT);

  // ---------------- layer 1 (H=4, D=32) ----------------
  k_zero_f32<<<cdiv(NF, NT), NT, 0, stream>>>(acc, NF);
  for (int r = 0; r < N_REL; ++r) {
    const int* src = edge_src + (size_t)r * N_EDGES;
    const int* dst = edge_dst + (size_t)r * N_EDGES;
    k_gemm_bf16<<<gemmGrid, NT, 0, stream>>>(hb, W1p + (size_t)r * WPACK_DW_PER_REL, feat);
    k_attn_proj<<<cdiv((long long)N_NODES * 4 * 32, NT), NT, 0, stream>>>(
        feat, al1 + r * 128, ar1 + r * 128, el, er, 4, 32);
    k_init_nodes<<<cdiv(N_NODES * 4, NT), NT, 0, stream>>>(nmax, nsum, N_NODES * 4);
    k_edge_max<<<cdiv(N_EDGES * 4, NT), NT, 0, stream>>>(src, dst, el, er, nmax, 4);
    k_edge_sum<<<cdiv(N_EDGES * 4, NT), NT, 0, stream>>>(src, dst, el, er, nmax, nsum, 4);
    k_edge_aggregate<<<aggGrid, NT, 0, stream>>>(src, dst, el, er, nmax, nsum, feat, acc, 4, 5);
  }
  k_finalize1<<<cdiv(NF, NT), NT, 0, stream>>>(acc, x, b1, h, hb);

  // ---------------- layer 2 (H=1, D=128) ----------------
  k_zero_f32<<<cdiv(NF, NT), NT, 0, stream>>>(acc, NF);
  for (int r = 0; r < N_REL; ++r) {
    const int* src = edge_src + (size_t)r * N_EDGES;
    const int* dst = edge_dst + (size_t)r * N_EDGES;
    k_gemm_bf16<<<gemmGrid, NT, 0, stream>>>(hb, W2p + (size_t)r * WPACK_DW_PER_REL, feat);
    k_attn_proj<<<cdiv((long long)N_NODES * 32, NT), NT, 0, stream>>>(
        feat, al2 + r * 128, ar2 + r * 128, el, er, 1, 128);
    k_init_nodes<<<cdiv(N_NODES, NT), NT, 0, stream>>>(nmax, nsum, N_NODES);
    k_edge_max<<<cdiv(N_EDGES, NT), NT, 0, stream>>>(src, dst, el, er, nmax, 1);
    k_edge_sum<<<cdiv(N_EDGES, NT), NT, 0, stream>>>(src, dst, el, er, nmax, nsum, 1);
    k_edge_aggregate<<<aggGrid, NT, 0, stream>>>(src, dst, el, er, nmax, nsum, feat, acc, 1, 7);
  }
  k_finalize2<<<cdiv(NF, NT), NT, 0, stream>>>(acc, h, b2, out);
}
